// MixtureOfExperts_14860586844770
// MI455X (gfx1250) — matrix operand order repaired and checked by reading, not compiled, not run on hardware
//
#include <hip/hip_runtime.h>
#include <cstdint>
#include <cstddef>

// ---------------------------------------------------------------------------
// MoE (top-2 of 8 experts), D=1024, F=4096, N=8192.  Routed bf16 WMMA GEMMs
// with double-buffered LDS and CDNA5 async global->LDS DMA.
// Workspace (assumes ws_size >= ~294 MB): counters | eidx | gates | rowmap |
// x_bf16[8192][1024] | W1t[8][4096][1024] | W2t[8][1024][4096] | h[17408][4096]
// ---------------------------------------------------------------------------

#define DM     1024
#define DF     4096
#define NTOK   8192
#define NE     8
#define ROWCAP 17408   // 16384 + 8*128 padding, 128-aligned segments

#define BM 128
#define BN 128
#define BK 32
#define AS 40          // LDS row stride in bf16 elements (32 + 8 pad), 80B rows

typedef __bf16 bf16_t;
typedef bf16_t v16bf __attribute__((ext_vector_type(16)));
typedef float  v8f   __attribute__((ext_vector_type(8)));

union FragU { v16bf v; uint4 q[2]; };

__device__ __forceinline__ unsigned short f2bf(float f) {
  unsigned int u = __float_as_uint(f);
  u += 0x7FFFu + ((u >> 16) & 1u);          // round-to-nearest-even
  return (unsigned short)(u >> 16);
}

// ------------------------- CDNA5 async global->LDS staging (with fallback) --
#if __has_builtin(__builtin_amdgcn_global_load_async_to_lds_b128)
#define HAVE_ASYNC_LDS 1
typedef int v4i_ __attribute__((__vector_size__(16)));
typedef __attribute__((address_space(1))) v4i_ gv4i;   // global
typedef __attribute__((address_space(3))) v4i_ lv4i;   // LDS
#else
#define HAVE_ASYNC_LDS 0
#endif

// copy 32 contiguous bytes (16 bf16) global -> LDS
__device__ __forceinline__ void copy32B(const unsigned short* g, unsigned short* l) {
#if HAVE_ASYNC_LDS
  __builtin_amdgcn_global_load_async_to_lds_b128((gv4i*)g, (lv4i*)l, 0, 0);
  __builtin_amdgcn_global_load_async_to_lds_b128((gv4i*)(g + 8), (lv4i*)(l + 8), 0, 0);
#else
  const uint4* s4 = (const uint4*)g;
  uint4* d4 = (uint4*)l;
  d4[0] = s4[0]; d4[1] = s4[1];
#endif
}

__device__ __forceinline__ void wait_stage() {
#if HAVE_ASYNC_LDS
#if __has_builtin(__builtin_amdgcn_s_wait_asynccnt)
  __builtin_amdgcn_s_wait_asynccnt(0);
#else
  asm volatile("s_wait_asynccnt 0" ::: "memory");
#endif
#endif
}

// ---------------------------------------------------------------- zero / misc
__global__ __launch_bounds__(256) void k_zero(float* __restrict__ out, int n4,
                                              int* __restrict__ misc) {
  int i = blockIdx.x * 256 + threadIdx.x;
  if (i < n4) ((float4*)out)[i] = make_float4(0.f, 0.f, 0.f, 0.f);
  if (blockIdx.x == 0 && threadIdx.x < 64) misc[threadIdx.x] = 0;
}

// ------------------------------------------------------------- f32 -> bf16
__global__ __launch_bounds__(256) void k_cvt_x(const float* __restrict__ x,
                                               unsigned short* __restrict__ xb) {
  int i = blockIdx.x * 256 + threadIdx.x;            // handles 4 elements
  float4 v = ((const float4*)x)[i];
  unsigned long long p = (unsigned long long)f2bf(v.x)
                       | ((unsigned long long)f2bf(v.y) << 16)
                       | ((unsigned long long)f2bf(v.z) << 32)
                       | ((unsigned long long)f2bf(v.w) << 48);
  ((unsigned long long*)xb)[i] = p;
}

// W1t[e][f][k] = W1[e][k][f]
__global__ __launch_bounds__(256) void k_cvt_w1(const float* __restrict__ W1,
                                                unsigned short* __restrict__ w1t) {
  long long i = (long long)blockIdx.x * 256 + threadIdx.x;
  long long base = i * 4;
  int k = (int)(base & (DM - 1));
  long long r = base >> 10;
  int f = (int)(r & (DF - 1));
  int e = (int)(r >> 12);
  unsigned long long p = 0;
#pragma unroll
  for (int t = 0; t < 4; ++t) {
    float v = W1[((long long)(e * DM + k + t)) * DF + f];
    p |= ((unsigned long long)f2bf(v)) << (16 * t);
  }
  ((unsigned long long*)w1t)[i] = p;
}

// W2t[e][d][f] = W2[e][f][d]
__global__ __launch_bounds__(256) void k_cvt_w2(const float* __restrict__ W2,
                                                unsigned short* __restrict__ w2t) {
  long long i = (long long)blockIdx.x * 256 + threadIdx.x;
  long long base = i * 4;
  int f = (int)(base & (DF - 1));
  long long r = base >> 12;
  int d = (int)(r & (DM - 1));
  int e = (int)(r >> 10);
  unsigned long long p = 0;
#pragma unroll
  for (int t = 0; t < 4; ++t) {
    float v = W2[((long long)(e * DF + f + t)) * DM + d];
    p |= ((unsigned long long)f2bf(v)) << (16 * t);
  }
  ((unsigned long long*)w2t)[i] = p;
}

// ------------------------------------------------------------------- gating
__global__ __launch_bounds__(256) void k_gate(const float* __restrict__ x,
                                              const float* __restrict__ Wg,
                                              const float* __restrict__ bg,
                                              int* __restrict__ counts,
                                              int* __restrict__ eidx,
                                              float* __restrict__ gates) {
  __shared__ float sWg[DM * NE];                     // 32 KB of 320 KB LDS
  for (int i = threadIdx.x; i < DM * NE; i += 256) sWg[i] = Wg[i];
  __syncthreads();
  int n = blockIdx.x * 256 + threadIdx.x;
  float acc[NE];
#pragma unroll
  for (int e = 0; e < NE; ++e) acc[e] = bg[e];
  const float* xr = x + (size_t)n * DM;
  for (int d = 0; d < DM; ++d) {
    float xv = xr[d];
#pragma unroll
    for (int e = 0; e < NE; ++e) acc[e] = fmaf(xv, sWg[d * NE + e], acc[e]);
  }
  int i0 = 0; float v0 = acc[0];
#pragma unroll
  for (int e = 1; e < NE; ++e) if (acc[e] > v0) { v0 = acc[e]; i0 = e; }
  int i1 = -1; float v1 = -3.4e38f;
#pragma unroll
  for (int e = 0; e < NE; ++e)
    if (e != i0 && acc[e] > v1) { v1 = acc[e]; i1 = e; }
  float g0 = 1.f / (1.f + __expf(v1 - v0));
  float g1 = 1.f - g0;
  eidx[n * 2 + 0] = i0; eidx[n * 2 + 1] = i1;
  gates[n * 2 + 0] = g0; gates[n * 2 + 1] = g1;
  atomicAdd(&counts[i0], 1);
  atomicAdd(&counts[i1], 1);
}

// ---------------------------------------------------- scan / scatter routing
__global__ void k_scan(const int* __restrict__ counts,
                       int* __restrict__ seg_start, int* __restrict__ seg_rows) {
  if (threadIdx.x == 0 && blockIdx.x == 0) {
    int run = 0;
    for (int e = 0; e < NE; ++e) {
      seg_start[e] = run;
      int a = (counts[e] + (BM - 1)) & ~(BM - 1);
      seg_rows[e] = a;
      run += a;
    }
  }
}

__global__ __launch_bounds__(256) void k_rowmap_init(int* __restrict__ rowmap) {
  int i = blockIdx.x * 256 + threadIdx.x;
  if (i < ROWCAP) rowmap[i] = -1;
}

__global__ __launch_bounds__(256) void k_scatter(const int* __restrict__ eidx,
                                                 const int* __restrict__ seg_start,
                                                 int* __restrict__ cursors,
                                                 int* __restrict__ rowmap) {
  int i = blockIdx.x * 256 + threadIdx.x;   // (token,slot) pair id
  int e = eidx[i];
  int p = atomicAdd(&cursors[e], 1);
  rowmap[seg_start[e] + p] = i;             // encodes token=(i>>1), slot=(i&1)
}

// --------------------------------------------------------------- FFN pass 1
// h[row][f] = relu( x[tok(row)] @ W1[e] + b1[e] ), row in compacted segment.
__global__ __launch_bounds__(256)
void k_ffn1(const unsigned short* __restrict__ xb,
            const unsigned short* __restrict__ w1t,
            const float* __restrict__ b1,
            const int* __restrict__ rowmap,
            const int* __restrict__ seg_start,
            const int* __restrict__ seg_rows,
            unsigned short* __restrict__ h) {
  const int e = blockIdx.z;
  const int mt = blockIdx.y;
  if (mt * BM >= seg_rows[e]) return;
  const int f0 = blockIdx.x * BN;
  const int rowbase = seg_start[e] + mt * BM;

  __shared__ unsigned short sA[2][BM * AS];
  __shared__ unsigned short sB[2][BN * AS];
  __shared__ int tokRow[BM];

  const int tid = threadIdx.x;
  if (tid < BM) {
    int mv = rowmap[rowbase + tid];
    tokRow[tid] = (mv >= 0) ? (mv >> 1) : 0;
  }
  __syncthreads();

  const int lane = tid & 31, wave = tid >> 5;
  const int wm = wave & 1, wn = wave >> 1;     // 2 x 4 wave grid
  const int lm = lane & 15, kh = lane >> 4;
  const int srow = tid >> 1, shalf = tid & 1;  // staging: 128 rows x 2 halves

  v8f acc[4][2];
#pragma unroll
  for (int i = 0; i < 4; ++i)
#pragma unroll
    for (int j = 0; j < 2; ++j)
#pragma unroll
      for (int t = 0; t < 8; ++t) acc[i][j][t] = 0.f;

  const unsigned short* gA = xb + (size_t)tokRow[srow] * DM + shalf * 16;
  const unsigned short* gB =
      w1t + (size_t)e * DF * DM + (size_t)(f0 + srow) * DM + shalf * 16;
  const int lofs = srow * AS + shalf * 16;

  // prologue: stage K-tile 0 into buffer 0
  copy32B(gA, &sA[0][lofs]);
  copy32B(gB, &sB[0][lofs]);
  wait_stage();
  __syncthreads();

  const int nIter = DM / BK;
  for (int it = 0; it < nIter; ++it) {
    const int cur = it & 1, nxt = cur ^ 1;
    if (it + 1 < nIter) {                      // async-prefetch next K tile
      copy32B(gA + (it + 1) * BK, &sA[nxt][lofs]);
      copy32B(gB + (it + 1) * BK, &sB[nxt][lofs]);
    }

    FragU bfr[2];
#pragma unroll
    for (int j = 0; j < 2; ++j) {     // B: lane<16 -> K{0..7,16..23}, else +8
      int col = wn * 32 + j * 16 + lm;
      bfr[j].q[0] = *(const uint4*)&sB[cur][col * AS + kh * 8];
      bfr[j].q[1] = *(const uint4*)&sB[cur][col * AS + 16 + kh * 8];
    }
#pragma unroll
    for (int i = 0; i < 4; ++i) {     // A: lane<16 -> K{0..7,16..23}, else +8
      FragU afr;
      int row = wm * 64 + i * 16 + lm;
      afr.q[0] = *(const uint4*)&sA[cur][row * AS + kh * 8];
      afr.q[1] = *(const uint4*)&sA[cur][row * AS + 16 + kh * 8];
#pragma unroll
      for (int j = 0; j < 2; ++j)
        acc[i][j] = __builtin_amdgcn_wmma_f32_16x16x32_bf16(
            false, afr.v, false, bfr[j].v, (short)0, acc[i][j], false, false);
    }
    wait_stage();                     // my async writes to buf[nxt] landed
    __syncthreads();                  // everyone's writes landed, reads done
  }

  const float* b1e = b1 + e * DF;
#pragma unroll
  for (int i = 0; i < 4; ++i)
#pragma unroll
    for (int j = 0; j < 2; ++j)
#pragma unroll
      for (int r = 0; r < 8; ++r) {
        int ml = wm * 64 + i * 16 + r + (kh ? 8 : 0);
        int f  = f0 + wn * 32 + j * 16 + lm;
        float v = acc[i][j][r] + b1e[f];
        v = v > 0.f ? v : 0.f;
        h[(size_t)(rowbase + ml) * DF + f] = f2bf(v);
      }
}

// --------------------------------------------------------------- FFN pass 2
// out[tok] += gate * ( h[row] @ W2[e] + b2[e] )   (float atomics; 2 hits/token)
__global__ __launch_bounds__(256)
void k_ffn2(const unsigned short* __restrict__ h,
            const unsigned short* __restrict__ w2t,
            const float* __restrict__ b2,
            const int* __restrict__ rowmap,
            const int* __restrict__ seg_start,
            const int* __restrict__ seg_rows,
            const float* __restrict__ gates,
            float* __restrict__ out) {
  const int e = blockIdx.z;
  const int mt = blockIdx.y;
  if (mt * BM >= seg_rows[e]) return;
  const int d0 = blockIdx.x * BN;
  const int rowbase = seg_start[e] + mt * BM;

  __shared__ unsigned short sA[2][BM * AS];
  __shared__ unsigned short sB[2][BN * AS];
  __shared__ int mvRow[BM];

  const int tid = threadIdx.x;
  if (tid < BM) mvRow[tid] = rowmap[rowbase + tid];
  __syncthreads();

  const int lane = tid & 31, wave = tid >> 5;
  const int wm = wave & 1, wn = wave >> 1;
  const int lm = lane & 15, kh = lane >> 4;
  const int srow = tid >> 1, shalf = tid & 1;

  v8f acc[4][2];
#pragma unroll
  for (int i = 0; i < 4; ++i)
#pragma unroll
    for (int j = 0; j < 2; ++j)
#pragma unroll
      for (int t = 0; t < 8; ++t) acc[i][j][t] = 0.f;

  const unsigned short* gA = h + (size_t)(rowbase + srow) * DF + shalf * 16;
  const unsigned short* gB =
      w2t + (size_t)e * DM * DF + (size_t)(d0 + srow) * DF + shalf * 16;
  const int lofs = srow * AS + shalf * 16;

  copy32B(gA, &sA[0][lofs]);
  copy32B(gB, &sB[0][lofs]);
  wait_stage();
  __syncthreads();

  const int nIter = DF / BK;
  for (int it = 0; it < nIter; ++it) {
    const int cur = it & 1, nxt = cur ^ 1;
    if (it + 1 < nIter) {
      copy32B(gA + (it + 1) * BK, &sA[nxt][lofs]);
      copy32B(gB + (it + 1) * BK, &sB[nxt][lofs]);
    }

    FragU bfr[2];
#pragma unroll
    for (int j = 0; j < 2; ++j) {
      int col = wn * 32 + j * 16 + lm;
      bfr[j].q[0] = *(const uint4*)&sB[cur][col * AS + kh * 8];
      bfr[j].q[1] = *(const uint4*)&sB[cur][col * AS + 16 + kh * 8];
    }
#pragma unroll
    for (int i = 0; i < 4; ++i) {
      FragU afr;
      int row = wm * 64 + i * 16 + lm;
      afr.q[0] = *(const uint4*)&sA[cur][row * AS + kh * 8];
      afr.q[1] = *(const uint4*)&sA[cur][row * AS + 16 + kh * 8];
#pragma unroll
      for (int j = 0; j < 2; ++j)
        acc[i][j] = __builtin_amdgcn_wmma_f32_16x16x32_bf16(
            false, afr.v, false, bfr[j].v, (short)0, acc[i][j], false, false);
    }
    wait_stage();
    __syncthreads();
  }

  const float* b2e = b2 + e * DM;
#pragma unroll
  for (int i = 0; i < 4; ++i)
#pragma unroll
    for (int j = 0; j < 2; ++j)
#pragma unroll
      for (int r = 0; r < 8; ++r) {
        int ml = wm * 64 + i * 16 + r + (kh ? 8 : 0);
        int mv = mvRow[ml];
        if (mv >= 0) {
          int d = d0 + wn * 32 + j * 16 + lm;
          float v = (acc[i][j][r] + b2e[d]) * gates[mv];
          unsafeAtomicAdd(&out[(size_t)(mv >> 1) * DM + d], v);
        }
      }
}

// ------------------------------------------------------------------- launch
extern "C" void kernel_launch(void* const* d_in, const int* in_sizes, int n_in,
                              void* d_out, int out_size, void* d_ws, size_t ws_size,
                              hipStream_t stream) {
  const float* x  = (const float*)d_in[0];
  const float* Wg = (const float*)d_in[1];
  const float* bg = (const float*)d_in[2];
  const float* W1 = (const float*)d_in[3];
  const float* b1 = (const float*)d_in[4];
  const float* W2 = (const float*)d_in[5];
  const float* b2 = (const float*)d_in[6];
  float* out = (float*)d_out;

  uint8_t* ws = (uint8_t*)d_ws;
  int*   misc      = (int*)(ws + 0);          // counts/cursors/seg_start/seg_rows
  int*   counts    = (int*)(ws + 0);
  int*   cursors   = (int*)(ws + 64);
  int*   seg_start = (int*)(ws + 128);
  int*   seg_rows  = (int*)(ws + 192);
  int*   eidx      = (int*)(ws + 256);
  float* gates     = (float*)(ws + 256 + 65536);
  int*   rowmap    = (int*)(ws + 256 + 131072);
  unsigned short* xb  = (unsigned short*)(ws + 200960);
  unsigned short* w1t = xb  + (size_t)NTOK * DM;
  unsigned short* w2t = w1t + (size_t)NE * DF * DM;
  unsigned short* h   = w2t + (size_t)NE * DM * DF;

  // 1) zero output + routing counters
  k_zero<<<dim3(NTOK * DM / (256 * 4)), 256, 0, stream>>>(out, NTOK * DM / 4, misc);
  // 2) precision conversion (+ weight transpose to K-contiguous layouts)
  k_cvt_x <<<dim3(NTOK * DM / (256 * 4)), 256, 0, stream>>>(x, xb);
  k_cvt_w1<<<dim3(NE * DF * DM / (256 * 4)), 256, 0, stream>>>(W1, w1t);
  k_cvt_w2<<<dim3(NE * DF * DM / (256 * 4)), 256, 0, stream>>>(W2, w2t);
  // 3) routing
  k_gate<<<dim3(NTOK / 256), 256, 0, stream>>>(x, Wg, bg, counts, eidx, gates);
  k_scan<<<dim3(1), 32, 0, stream>>>(counts, seg_start, seg_rows);
  k_rowmap_init<<<dim3((ROWCAP + 255) / 256), 256, 0, stream>>>(rowmap);
  k_scatter<<<dim3(NTOK * 2 / 256), 256, 0, stream>>>(eidx, seg_start, cursors, rowmap);
  // 4) routed GEMMs (worst-case grids; blocks early-exit past segment end)
  k_ffn1<<<dim3(DF / BN, 2 * NTOK / BM, NE), 256, 0, stream>>>(
      xb, w1t, b1, rowmap, seg_start, seg_rows, h);
  k_ffn2<<<dim3(DM / BN, 2 * NTOK / BM, NE), 256, 0, stream>>>(
      h, w2t, b2, rowmap, seg_start, seg_rows, gates, out);
}
